// GraphAttentionLayer_7421703487713
// MI455X (gfx1250) — matrix operand order repaired and checked by reading, not compiled, not run on hardware
//
#include <hip/hip_runtime.h>
#include <hip/hip_bf16.h>

// ---------------------------------------------------------------------------
// GAT layer for MI455X (gfx1250, wave32, WMMA).
// Heavy math runs on v_wmma_f32_16x16x32_bf16 (f32 accumulate).
// Global->LDS staging uses CDNA5 async-LDS loads (ASYNCcnt) with double
// buffering when the builtins are available; sync b128 staging otherwise.
// ---------------------------------------------------------------------------

typedef __attribute__((ext_vector_type(16))) __bf16 v16bf;
typedef __attribute__((ext_vector_type(8)))  __bf16 v8bf;
typedef __attribute__((ext_vector_type(8)))  float  v8f;
typedef __attribute__((ext_vector_type(4)))  int    v4i;

#define NB_  8        // batch
#define NN_  1024     // nodes
#define NIN_ 256      // IN
#define ND_  768      // 3*IN  (concat feature dim)
#define NOUT_ 256     // OUT
#define ALPHA_ 0.2f
#define NEGV_  -9.0e15f
#define TP_   40      // padded LDS tile pitch in bf16 (80B: conflict-free frags)

#if defined(__AMDGCN__) && \
    __has_builtin(__builtin_amdgcn_global_load_async_to_lds_b128) && \
    __has_builtin(__builtin_amdgcn_s_wait_asynccnt)
#define USE_ASYNC 1
typedef __attribute__((address_space(1))) v4i glob_v4i;   // global int4
typedef __attribute__((address_space(3))) v4i lds_v4i;    // LDS int4
#else
#define USE_ASYNC 0
#endif

// ---------------------------------------------------------------------------
// Stage one 128x32 A-tile + 64x32 Bt-tile (bf16) into padded LDS buffers.
// ---------------------------------------------------------------------------
__device__ __forceinline__ void stage_tile(const __bf16* __restrict__ A, int lda, int m0,
                                           const __bf16* __restrict__ Bt, int ldb, int n0,
                                           int k0, __bf16* sA, __bf16* sB)
{
    const int tid = threadIdx.x;
#if USE_ASYNC
    // 3 async b128 ops per thread: ASYNCcnt-tracked, no VGPR round-trip.
#pragma unroll
    for (int c = tid; c < 512; c += 256) {
        const int row = c >> 2;
        const int ch  = c & 3;
        __builtin_amdgcn_global_load_async_to_lds_b128(
            (glob_v4i*)(A + (size_t)(m0 + row) * lda + k0 + ch * 8),
            (lds_v4i*)(sA + row * TP_ + ch * 8), 0, 0);
    }
    {
        const int row = tid >> 2;
        const int ch  = tid & 3;
        __builtin_amdgcn_global_load_async_to_lds_b128(
            (glob_v4i*)(Bt + (size_t)(n0 + row) * ldb + k0 + ch * 8),
            (lds_v4i*)(sB + row * TP_ + ch * 8), 0, 0);
    }
#else
#pragma unroll
    for (int c = tid; c < 512; c += 256) {
        const int row = c >> 2;
        const int ch  = c & 3;
        *(uint4*)(sA + row * TP_ + ch * 8) =
            *(const uint4*)(A + (size_t)(m0 + row) * lda + k0 + ch * 8);
    }
    {
        const int row = tid >> 2;
        const int ch  = tid & 3;
        *(uint4*)(sB + row * TP_ + ch * 8) =
            *(const uint4*)(Bt + (size_t)(n0 + row) * ldb + k0 + ch * 8);
    }
    __builtin_prefetch(A + (size_t)(m0 + (tid & 127)) * lda + k0 + 32, 0, 1);
#endif
}

// ---------------------------------------------------------------------------
// Shared WMMA GEMM core: C[128x64] tile of  A(MxK, row-major) * Bt(NxK)^T
// 256 threads = 8 waves; each wave owns a 32x32 sub-tile (2x2 WMMA frags).
// Double-buffered LDS: tile k+1 streams in while tile k feeds the WMMAs.
// ---------------------------------------------------------------------------
__device__ __forceinline__ void gemm_128x64(const __bf16* __restrict__ A, int lda,
                                            const __bf16* __restrict__ Bt, int ldb,
                                            int K, int m0, int n0,
                                            v8f acc[2][2])
{
    __shared__ __bf16 sA[2][128 * TP_];   // 2 x 10 KB
    __shared__ __bf16 sB[2][64 * TP_];    // 2 x  5 KB

    const int tid  = threadIdx.x;
    const int lane = tid & 31;
    const int wave = tid >> 5;
    const int wm   = (wave & 3) * 32;   // wave m-offset in block tile
    const int wn   = (wave >> 2) * 32;  // wave n-offset in block tile
    const int l15  = lane & 15;
    const int hi   = lane >> 4;         // 0/1: which half of the K-slice

#pragma unroll
    for (int fm = 0; fm < 2; ++fm)
#pragma unroll
        for (int fn = 0; fn < 2; ++fn) {
            v8f z = {0.f, 0.f, 0.f, 0.f, 0.f, 0.f, 0.f, 0.f};
            acc[fm][fn] = z;
        }

    // prologue: stage first tile
    stage_tile(A, lda, m0, Bt, ldb, n0, 0, sA[0], sB[0]);
#if USE_ASYNC
    __builtin_amdgcn_s_wait_asynccnt(0);
#endif
    __syncthreads();

    for (int k0 = 0; k0 < K; k0 += 32) {
        const int cur = (k0 >> 5) & 1;
        // stream next tile into the other buffer while we compute this one
        if (k0 + 32 < K)
            stage_tile(A, lda, m0, Bt, ldb, n0, k0 + 32, sA[cur ^ 1], sB[cur ^ 1]);

        const __bf16* a_ = sA[cur];
        const __bf16* b_ = sB[cur];
        v16bf af[2], bfv[2];
#pragma unroll
        for (int f = 0; f < 2; ++f) {
            // A 16x32 bf16 frag: lane m=l15, halves at K = kb..kb+7, kb+16..kb+23
            const int m = wm + f * 16 + l15;
            v8bf alo = *(const v8bf*)(a_ + m * TP_ + hi * 8);
            v8bf ahi = *(const v8bf*)(a_ + m * TP_ + hi * 8 + 16);
            af[f] = __builtin_shufflevector(alo, ahi,
                     0, 1, 2, 3, 4, 5, 6, 7, 8, 9, 10, 11, 12, 13, 14, 15);
            // B 32x16 bf16 frag: lane n=l15, halves at K = kb..kb+7, kb+16..kb+23
            const int n = wn + f * 16 + l15;
            v8bf blo = *(const v8bf*)(b_ + n * TP_ + hi * 8);
            v8bf bhi = *(const v8bf*)(b_ + n * TP_ + hi * 8 + 16);
            bfv[f] = __builtin_shufflevector(blo, bhi,
                     0, 1, 2, 3, 4, 5, 6, 7, 8, 9, 10, 11, 12, 13, 14, 15);
        }

#pragma unroll
        for (int fm = 0; fm < 2; ++fm)
#pragma unroll
            for (int fn = 0; fn < 2; ++fn)
                acc[fm][fn] = __builtin_amdgcn_wmma_f32_16x16x32_bf16(
                    false, af[fm], false, bfv[fn], (short)0, acc[fm][fn],
                    false, false);

#if USE_ASYNC
        __builtin_amdgcn_s_wait_asynccnt(0);   // next buffer resident
#endif
        __syncthreads();                       // all waves done with cur buffer
    }
}

// ---------------------------------------------------------------------------
// Prep kernels
// ---------------------------------------------------------------------------
__global__ void cvt_bf16_kernel(const float* __restrict__ s, __bf16* __restrict__ d, int n)
{
    int i = blockIdx.x * blockDim.x + threadIdx.x;
    if (i < n) d[i] = (__bf16)s[i];
}

// x (B, IN, N) -> inp_bf16 (B, N, IN), LDS 32x32 tiled transpose
__global__ void __launch_bounds__(256)
transpose_x_kernel(const float* __restrict__ x, __bf16* __restrict__ inpb)
{
    __shared__ float tile[32][33];
    const int b  = blockIdx.z;
    const int n0 = blockIdx.x * 32;
    const int k0 = blockIdx.y * 32;
    const int tx = threadIdx.x;      // 0..31
    const int ty = threadIdx.y;      // 0..7
    const float* xb = x + (size_t)b * NIN_ * NN_;
#pragma unroll
    for (int r = 0; r < 32; r += 8)
        tile[ty + r][tx] = xb[(size_t)(k0 + ty + r) * NN_ + n0 + tx];
    __syncthreads();
    __bf16* ob = inpb + (size_t)b * NN_ * NIN_;
#pragma unroll
    for (int r = 0; r < 32; r += 8)
        ob[(size_t)(n0 + ty + r) * NIN_ + k0 + tx] = (__bf16)tile[tx][ty + r];
}

// ax (B, 2*IN, N) copied (bf16) straight into rows 256..767 of hT (B, 768, N)
__global__ void ax_to_hT_kernel(const float* __restrict__ ax, __bf16* __restrict__ hT)
{
    const int b = blockIdx.y;
    const int i = blockIdx.x * 256 + threadIdx.x;   // 0 .. 2*IN*N-1
    hT[(size_t)b * ND_ * NN_ + (size_t)NIN_ * NN_ + i] =
        (__bf16)ax[(size_t)b * 2 * NIN_ * NN_ + i];
}

// ---------------------------------------------------------------------------
// GEMM 1:  h[n][j] = sum_k inp[n][k] * WQ_w[j][k] + WQ_b[j]
// stored transposed: hT[b][j][n]  (contiguous v8bf store per fragment column)
// ---------------------------------------------------------------------------
__global__ void __launch_bounds__(256)
gemm_h_kernel(const __bf16* __restrict__ inpb, const __bf16* __restrict__ wqb,
              const float* __restrict__ wq_bias, __bf16* __restrict__ hT)
{
    const int b  = blockIdx.z;
    const int m0 = blockIdx.x * 128;   // node index
    const int n0 = blockIdx.y * 64;    // output feature j
    v8f acc[2][2];
    gemm_128x64(inpb + (size_t)b * NN_ * NIN_, NIN_, wqb, NIN_, NIN_, m0, n0, acc);

    const int lane = threadIdx.x & 31, wave = threadIdx.x >> 5;
    const int wm = (wave & 3) * 32, wn = (wave >> 2) * 32;
    const int l15 = lane & 15, hi = lane >> 4;
    __bf16* hb = hT + (size_t)b * ND_ * NN_;
#pragma unroll
    for (int fm = 0; fm < 2; ++fm)
#pragma unroll
        for (int fn = 0; fn < 2; ++fn) {
            const int j  = n0 + wn + fn * 16 + l15;
            const int mb = m0 + wm + fm * 16 + hi * 8;   // 8 consecutive nodes
            const float bias = wq_bias[j];
            v8bf pk;
#pragma unroll
            for (int v = 0; v < 8; ++v)
                pk[v] = (__bf16)(acc[fm][fn][v] + bias);
            *(v8bf*)(hb + (size_t)j * NN_ + mb) = pk;
        }
}

// ---------------------------------------------------------------------------
// f1/f2: column reductions over hT (coalesced across nodes)
// ---------------------------------------------------------------------------
__global__ void f_kernel(const __bf16* __restrict__ hT, const float* __restrict__ a,
                         float* __restrict__ f1, float* __restrict__ f2)
{
    const int t = blockIdx.x * blockDim.x + threadIdx.x;   // 0..8191
    const int b = t >> 10, i = t & 1023;
    const __bf16* hb = hT + (size_t)b * ND_ * NN_;
    float s1 = 0.f, s2 = 0.f;
    for (int d = 0; d < ND_; ++d) {
        float hv = (float)hb[(size_t)d * NN_ + i];
        s1 += hv * a[d];
        s2 += hv * a[ND_ + d];
    }
    f1[t] = s1;
    f2[t] = s2;
}

// ---------------------------------------------------------------------------
// Attention row softmax (wave32 per row, online max/sum, adj mask, bf16 out)
// ---------------------------------------------------------------------------
__global__ void __launch_bounds__(256)
att_kernel(const float* __restrict__ f1, const float* __restrict__ f2,
           const int* __restrict__ adj, __bf16* __restrict__ att)
{
    const int wave = threadIdx.x >> 5;
    const int lane = threadIdx.x & 31;
    const int row  = blockIdx.x * 8 + wave;   // 0..8191
    const int b    = row >> 10;
    const float f1i = f1[row];
    const int*   adjrow = adj + (size_t)row * NN_;
    const float* f2b    = f2 + b * NN_;

    float m = -3.0e38f, s = 0.f;
    for (int j = lane; j < NN_; j += 32) {
        float e = f1i + f2b[j];
        e = e > 0.f ? e : ALPHA_ * e;
        if (adjrow[j] <= 0) e = NEGV_;
        if (e > m) { s = s * __expf(m - e) + 1.f; m = e; }
        else       { s += __expf(e - m); }
    }
#pragma unroll
    for (int off = 16; off > 0; off >>= 1) {
        float mo = __shfl_xor(m, off, 32);
        float so = __shfl_xor(s, off, 32);
        float mn = fmaxf(m, mo);
        s = s * __expf(m - mn) + so * __expf(mo - mn);
        m = mn;
    }
    const float inv = 1.f / s;
    __bf16* arow = att + (size_t)row * NN_;
    for (int j = lane; j < NN_; j += 32) {
        float e = f1i + f2b[j];
        e = e > 0.f ? e : ALPHA_ * e;
        if (adjrow[j] <= 0) e = NEGV_;
        arow[j] = (__bf16)(__expf(e - m) * inv);
    }
}

// ---------------------------------------------------------------------------
// GEMM 2: h'[i][d] = sum_j att[i][j] * h[j][d]   (Bt = hT, K-contiguous)
// ---------------------------------------------------------------------------
__global__ void __launch_bounds__(256)
gemm_ah_kernel(const __bf16* __restrict__ att, const __bf16* __restrict__ hT,
               __bf16* __restrict__ hp)
{
    const int b  = blockIdx.z;
    const int m0 = blockIdx.x * 128;  // node i
    const int n0 = blockIdx.y * 64;   // feature d
    v8f acc[2][2];
    gemm_128x64(att + (size_t)b * NN_ * NN_, NN_,
                hT  + (size_t)b * ND_ * NN_, NN_, NN_, m0, n0, acc);

    const int lane = threadIdx.x & 31, wave = threadIdx.x >> 5;
    const int wm = (wave & 3) * 32, wn = (wave >> 2) * 32;
    const int l15 = lane & 15, hi = lane >> 4;
    __bf16* hpb = hp + (size_t)b * NN_ * ND_;
#pragma unroll
    for (int fm = 0; fm < 2; ++fm)
#pragma unroll
        for (int fn = 0; fn < 2; ++fn) {
            const int d  = n0 + wn + fn * 16 + l15;
            const int mb = m0 + wm + fm * 16 + hi * 8;
#pragma unroll
            for (int v = 0; v < 8; ++v)
                hpb[(size_t)(mb + v) * ND_ + d] = (__bf16)acc[fm][fn][v];
        }
}

// ---------------------------------------------------------------------------
// GEMM 3: out = elu(h' * fc_w^T + fc_b), f32 output
// ---------------------------------------------------------------------------
__global__ void __launch_bounds__(256)
gemm_fc_kernel(const __bf16* __restrict__ hp, const __bf16* __restrict__ fcwb,
               const float* __restrict__ fc_bias, float* __restrict__ out)
{
    const int b  = blockIdx.z;
    const int m0 = blockIdx.x * 128;  // node
    const int n0 = blockIdx.y * 64;   // out feature
    v8f acc[2][2];
    gemm_128x64(hp + (size_t)b * NN_ * ND_, ND_, fcwb, ND_, ND_, m0, n0, acc);

    const int lane = threadIdx.x & 31, wave = threadIdx.x >> 5;
    const int wm = (wave & 3) * 32, wn = (wave >> 2) * 32;
    const int l15 = lane & 15, hi = lane >> 4;
    float* ob = out + (size_t)b * NN_ * NOUT_;
#pragma unroll
    for (int fm = 0; fm < 2; ++fm)
#pragma unroll
        for (int fn = 0; fn < 2; ++fn) {
            const int n  = n0 + wn + fn * 16 + l15;
            const int mb = m0 + wm + fm * 16 + hi * 8;
            const float bias = fc_bias[n];
#pragma unroll
            for (int v = 0; v < 8; ++v) {
                float val = acc[fm][fn][v] + bias;
                val = val > 0.f ? val : (expf(val) - 1.f);   // ELU, alpha=1
                ob[(size_t)(mb + v) * NOUT_ + n] = val;
            }
        }
}

// ---------------------------------------------------------------------------
// Host launcher
// ---------------------------------------------------------------------------
extern "C" void kernel_launch(void* const* d_in, const int* in_sizes, int n_in,
                              void* d_out, int out_size, void* d_ws, size_t ws_size,
                              hipStream_t stream)
{
    (void)in_sizes; (void)n_in; (void)out_size; (void)ws_size;

    const float* x    = (const float*)d_in[0];   // (B, IN, N)
    const float* ax   = (const float*)d_in[1];   // (B, 2IN, N)
    const int*   adj  = (const int*)  d_in[2];   // (B, N, N)
    const float* WQ_w = (const float*)d_in[3];   // (IN, IN)
    const float* WQ_b = (const float*)d_in[4];   // (IN,)
    const float* a    = (const float*)d_in[5];   // (6*IN, 1)
    const float* fc_w = (const float*)d_in[6];   // (OUT, 3IN)
    const float* fc_b = (const float*)d_in[7];   // (OUT,)
    float* out = (float*)d_out;

    // workspace carve-up (all offsets 256B-aligned)
    char* ws = (char*)d_ws;
    __bf16* wqb  = (__bf16*)ws;  ws += (size_t)NIN_ * NIN_ * 2;            // 128 KB
    __bf16* fcwb = (__bf16*)ws;  ws += (size_t)NOUT_ * ND_ * 2;            // 384 KB
    __bf16* inpb = (__bf16*)ws;  ws += (size_t)NB_ * NN_ * NIN_ * 2;       // 4 MB
    __bf16* hT   = (__bf16*)ws;  ws += (size_t)NB_ * ND_ * NN_ * 2;        // 12.6 MB
    float*  f1   = (float*) ws;  ws += (size_t)NB_ * NN_ * 4;              // 32 KB
    float*  f2   = (float*) ws;  ws += (size_t)NB_ * NN_ * 4;              // 32 KB
    __bf16* attw = (__bf16*)ws;  ws += (size_t)NB_ * NN_ * NN_ * 2;        // 16.8 MB
    __bf16* hp   = (__bf16*)ws;  ws += (size_t)NB_ * NN_ * ND_ * 2;        // 12.6 MB

    // --- prep: weight conversions, x transpose, ax -> hT rows 256..767
    cvt_bf16_kernel<<<(NIN_ * NIN_) / 256, 256, 0, stream>>>(WQ_w, wqb, NIN_ * NIN_);
    cvt_bf16_kernel<<<(NOUT_ * ND_) / 256, 256, 0, stream>>>(fc_w, fcwb, NOUT_ * ND_);
    transpose_x_kernel<<<dim3(NN_ / 32, NIN_ / 32, NB_), dim3(32, 8), 0, stream>>>(x, inpb);
    ax_to_hT_kernel<<<dim3((2 * NIN_ * NN_) / 256, NB_), 256, 0, stream>>>(ax, hT);

    // --- GEMM 1: h (transposed into hT rows 0..255)
    gemm_h_kernel<<<dim3(NN_ / 128, NIN_ / 64, NB_), 256, 0, stream>>>(inpb, wqb, WQ_b, hT);

    // --- attention logits + softmax
    f_kernel<<<(NB_ * NN_) / 256, 256, 0, stream>>>(hT, a, f1, f2);
    att_kernel<<<(NB_ * NN_) / 8, 256, 0, stream>>>(f1, f2, adj, attw);

    // --- GEMM 2: h' = att * h
    gemm_ah_kernel<<<dim3(NN_ / 128, ND_ / 64, NB_), 256, 0, stream>>>(attw, hT, hp);

    // --- GEMM 3: out = elu(h' * fc_w^T + fc_b)
    gemm_fc_kernel<<<dim3(NN_ / 128, NOUT_ / 64, NB_), 256, 0, stream>>>(hp, fcwb, fc_b, out);
}
